// Network_30382598652356
// MI455X (gfx1250) — compile-verified, matrix operand order checked by reading, not run on hardware
//
#include <hip/hip_runtime.h>
#include <hip/hip_bf16.h>

typedef __bf16 bf16;
typedef __attribute__((ext_vector_type(8)))  bf16  v8bf;
typedef __attribute__((ext_vector_type(16))) bf16  v16bf;
typedef __attribute__((ext_vector_type(8)))  float v8f;

#define NRAYS 4096
#define NSC 64
#define NSF 128

// ---- layer tables: 12 layers per net: pts0..7, alpha, feature, views, rgb ----
__device__ __host__ constexpr int WOFF[12] = {0,32768,98304,163840,229376,294912,393216,458752,524288,528384,593920,630784};
__device__ __host__ constexpr int KPAD[12] = {128,256,256,256,256,384,256,256,256,256,288,128};
__device__ __host__ constexpr int KREAL[12]= {127,256,256,256,256,383,256,256,256,256,283,128};
__device__ __host__ constexpr int NREAL[12]= {256,256,256,256,256,256,256,256,  1,256,128,  3};
__device__ __host__ constexpr int BOFF[12] = {0,256,512,768,1024,1280,1536,1792,2048,2064,2320,2448};
constexpr int WTOT = 632832;   // bf16 elements per net (transposed, padded)
constexpr int BTOT = 2464;     // f32 bias elements per net (padded)

struct PrepArgs { const float* W[24]; const float* Bv[24]; };

// Convert fp32 params -> bf16 W^T [Npad][Kpad] (contiguous-K rows for WMMA B frags),
// remapping K for padded input (L0), skip concat (L5: x_all|0|h), views concat.
__global__ void prep_params(PrepArgs a, bf16* wtc, bf16* wtf, float* bc, float* bf_) {
  long gid = (long)blockIdx.x * blockDim.x + threadIdx.x;
  const long PER = WTOT + BTOT;
  if (gid >= 2 * PER) return;
  int net = gid >= PER;
  long lid = gid - (long)net * PER;
  bf16* wt = net ? wtf : wtc;
  float* bo = net ? bf_ : bc;
  if (lid < WTOT) {
    int l = 0;
    for (int i = 1; i < 12; ++i) if (lid >= WOFF[i]) l = i;
    long rem = lid - WOFF[l];
    int k = (int)(rem % KPAD[l]);
    int n = (int)(rem / KPAD[l]);
    int sk;
    if (l == 5) sk = (k < 127) ? k : ((k == 127) ? -1 : k - 1);   // skip: [x_all(127) | 0 | h(256)]
    else        sk = (k < KREAL[l]) ? k : -1;
    float v = 0.f;
    if (sk >= 0 && n < NREAL[l]) v = a.W[net*12 + l][(long)sk * NREAL[l] + n];
    wt[lid] = (bf16)v;
  } else {
    long bid = lid - WTOT;
    int l = 0;
    for (int i = 1; i < 12; ++i) if (bid >= BOFF[i]) l = i;
    int n = (int)(bid - BOFF[l]);
    bo[bid] = (n < NREAL[l]) ? a.Bv[net*12 + l][n] : 0.f;
  }
}

// ---- in-block WMMA GEMM: 64 rows x N cols, activations in LDS, W^T from L2 ----
// MODE 0: LDS dst (+bias, opt relu, bf16). MODE 1: global raw rgb (n<3). MODE 2: global raw alpha (col 3).
template<int NT, int MODE, bool RELU>
__device__ __forceinline__ void wgemm(const bf16* __restrict__ WT, const float* __restrict__ bias,
                                      const bf16* __restrict__ src, int ldS,
                                      bf16* dst, int ldD, int dcol,
                                      float* __restrict__ graw, int K) {
  const int lane = threadIdx.x & 31;
  const int wv   = threadIdx.x >> 5;
  const int m0   = (wv & 3) << 4;      // 4 row bands of 16
  const int nh   = wv >> 2;            // 2 column halves
  const int l16  = lane & 15;
  const int lh   = lane >> 4;
  const int kb   = lh << 3;            // K sub-block: lanes 0-15 -> {kb, kb+16}, 16-31 -> {8+.., 24+..}
  const int nbase = nh * (NT * 16);

  v8f acc[NT];
  #pragma unroll
  for (int j = 0; j < NT; ++j)
    #pragma unroll
    for (int e = 0; e < 8; ++e) acc[j][e] = 0.f;

  const bf16* srow = src + (m0 + l16) * ldS;
  for (int k0 = 0; k0 < K; k0 += 32) {
    v8bf alo = *(const v8bf*)(srow + k0 + kb);
    v8bf ahi = *(const v8bf*)(srow + k0 + 16 + kb);
    v16bf af = __builtin_shufflevector(alo, ahi, 0,1,2,3,4,5,6,7,8,9,10,11,12,13,14,15);
    #pragma unroll
    for (int j = 0; j < NT; ++j) {
      const bf16* wrow = WT + (size_t)(nbase + j*16 + l16) * K;
      v8bf blo = *(const v8bf*)(wrow + k0 + kb);
      v8bf bhi = *(const v8bf*)(wrow + k0 + 16 + kb);
      v16bf bv = __builtin_shufflevector(blo, bhi, 0,1,2,3,4,5,6,7,8,9,10,11,12,13,14,15);
      acc[j] = __builtin_amdgcn_wmma_f32_16x16x32_bf16(false, af, false, bv, (short)0, acc[j], false, false);
    }
  }
  #pragma unroll
  for (int j = 0; j < NT; ++j) {
    int n = nbase + j*16 + l16;
    #pragma unroll
    for (int r = 0; r < 8; ++r) {
      int m = m0 + (lh << 3) + r;        // C/D layout: lanes 16-31 hold M=8..15
      float v = acc[j][r];
      if constexpr (MODE == 0) {
        v += bias[n];
        if constexpr (RELU) v = fmaxf(v, 0.f);
        dst[m * ldD + dcol + n] = (bf16)v;
      } else if constexpr (MODE == 1) {
        if (n < 3) graw[m * 4 + n] = v + bias[n];
      } else {
        if (n == 0) graw[m * 4 + 3] = v + bias[0];
      }
    }
  }
}

// ---- fully fused NeRF MLP: embed -> 8x256 MLP (skip@4) -> alpha/feature -> views -> rgb ----
__global__ void __launch_bounds__(256)
fused_mlp(const float* __restrict__ rays, const float* __restrict__ aud,
          const float* __restrict__ zf, const bf16* __restrict__ WT,
          const float* __restrict__ bias, float* __restrict__ raw, int S) {
  extern __shared__ bf16 smem[];
  bf16* sA = smem;                 // [64][400] : x_all | (skip h) | view emb
  bf16* sB = smem + 64 * 400;      // [64][272]
  bf16* sC = sB + 64 * 272;        // [64][272]
  const int base = blockIdx.x * 64;
  const int t = threadIdx.x;

  if (t < 64) {                    // positional embedding + audio
    int pidx = base + t;
    int ray = pidx / S;
    int s = pidx - ray * S;
    const float* rp = rays + ray * 11;
    float nr = rp[6], fr = rp[7];
    float z = (S == NSC) ? (nr + (fr - nr) * ((float)s / (float)(NSC - 1)))
                         : zf[ray * NSF + s];
    float p[3] = { rp[0] + rp[3]*z, rp[1] + rp[4]*z, rp[2] + rp[5]*z };
    bf16* row = sA + t * 400;
    row[0] = (bf16)p[0]; row[1] = (bf16)p[1]; row[2] = (bf16)p[2];
    int c = 3; float f = 1.f;
    for (int l = 0; l < 10; ++l) {
      row[c+0] = (bf16)__sinf(p[0]*f); row[c+1] = (bf16)__sinf(p[1]*f); row[c+2] = (bf16)__sinf(p[2]*f);
      row[c+3] = (bf16)__cosf(p[0]*f); row[c+4] = (bf16)__cosf(p[1]*f); row[c+5] = (bf16)__cosf(p[2]*f);
      c += 6; f *= 2.f;
    }
    for (int i = 0; i < 64; ++i) row[63 + i] = (bf16)aud[i];
    row[127] = (bf16)0.f;
  }
  __syncthreads();
  float* rawB = raw + (size_t)base * 4;
  wgemm<8,0,true >(WT+0,      bias+0,    sA,400, sB,272,0,   nullptr, 128); __syncthreads(); // L0
  wgemm<8,0,true >(WT+32768,  bias+256,  sB,272, sC,272,0,   nullptr, 256); __syncthreads(); // L1
  wgemm<8,0,true >(WT+98304,  bias+512,  sC,272, sB,272,0,   nullptr, 256); __syncthreads(); // L2
  wgemm<8,0,true >(WT+163840, bias+768,  sB,272, sC,272,0,   nullptr, 256); __syncthreads(); // L3
  wgemm<8,0,true >(WT+229376, bias+1024, sC,272, sA,400,128, nullptr, 256); __syncthreads(); // L4 -> skip buf
  wgemm<8,0,true >(WT+294912, bias+1280, sA,400, sB,272,0,   nullptr, 384); __syncthreads(); // L5 (skip)
  wgemm<8,0,true >(WT+393216, bias+1536, sB,272, sC,272,0,   nullptr, 256); __syncthreads(); // L6
  wgemm<8,0,true >(WT+458752, bias+1792, sC,272, sB,272,0,   nullptr, 256); __syncthreads(); // L7 -> h in sB
  wgemm<1,2,false>(WT+524288, bias+2048, sB,272, nullptr,0,0, rawB,   256);                  // alpha head
  if (t < 64) {                    // view-dir embedding into sA cols 256..287
    int pidx = base + t; int ray = pidx / S;
    const float* rp = rays + ray * 11;
    float d0 = rp[8], d1 = rp[9], d2 = rp[10];
    bf16* row = sA + t * 400 + 256;
    row[0] = (bf16)d0; row[1] = (bf16)d1; row[2] = (bf16)d2;
    int c = 3; float f = 1.f;
    for (int l = 0; l < 4; ++l) {
      row[c+0] = (bf16)__sinf(d0*f); row[c+1] = (bf16)__sinf(d1*f); row[c+2] = (bf16)__sinf(d2*f);
      row[c+3] = (bf16)__cosf(d0*f); row[c+4] = (bf16)__cosf(d1*f); row[c+5] = (bf16)__cosf(d2*f);
      c += 6; f *= 2.f;
    }
    for (int i = 27; i < 32; ++i) row[i] = (bf16)0.f;
  }
  wgemm<8,0,false>(WT+528384, bias+2064, sB,272, sA,400,0,   nullptr, 256); __syncthreads(); // feature (no relu)
  wgemm<4,0,true >(WT+593920, bias+2320, sA,400, sC,272,0,   nullptr, 288); __syncthreads(); // views
  wgemm<1,1,false>(WT+630784, bias+2448, sC,272, nullptr,0,0, rawB,   128);                  // rgb head
}

// ---- volume rendering: sequential transmittance per ray ----
__global__ void raw2out(const float* __restrict__ raw, const float* __restrict__ rays,
                        const float* __restrict__ bc, const float* __restrict__ zf,
                        float* __restrict__ rgb_out, float* __restrict__ w_out, int S) {
  int r = blockIdx.x * blockDim.x + threadIdx.x;
  if (r >= NRAYS) return;
  const float* rp = rays + r * 11;
  float nr = rp[6], fr = rp[7];
  float dn = sqrtf(rp[3]*rp[3] + rp[4]*rp[4] + rp[5]*rp[5]);
  float T = 1.f, a0 = 0.f, a1 = 0.f, a2 = 0.f;
  float zc = (S == NSC) ? nr : zf[r * NSF];
  for (int s = 0; s < S; ++s) {
    float zn = (s < S-1) ? ((S == NSC) ? (nr + (fr-nr) * ((float)(s+1)/(float)(NSC-1)))
                                       : zf[r * NSF + s + 1])
                         : zc;
    float dist = (s < S-1) ? (zn - zc) * dn : 1e10f * dn;
    const float* rw = raw + ((size_t)r * S + s) * 4;
    float c0, c1, c2;
    if (s == S-1) { c0 = bc[r*3]; c1 = bc[r*3+1]; c2 = bc[r*3+2]; }
    else {
      c0 = 1.f/(1.f + __expf(-rw[0]));
      c1 = 1.f/(1.f + __expf(-rw[1]));
      c2 = 1.f/(1.f + __expf(-rw[2]));
    }
    float al = 1.f - __expf(-fmaxf(rw[3], 0.f) * dist);
    float w = al * T;
    a0 += w*c0; a1 += w*c1; a2 += w*c2;
    if (w_out) w_out[r * S + s] = w;
    T *= (1.f - al + 1e-10f);
    zc = zn;
  }
  rgb_out[r*3] = a0; rgb_out[r*3+1] = a1; rgb_out[r*3+2] = a2;
}

// ---- inverse-CDF importance sampling + sorted merge (both inputs sorted) ----
__global__ void pdf_merge(const float* __restrict__ rays, const float* __restrict__ wbuf,
                          float* __restrict__ zf) {
  int r = blockIdx.x * blockDim.x + threadIdx.x;
  if (r >= NRAYS) return;
  float nr = rays[r*11+6], fr = rays[r*11+7];
  float z[NSC];
  for (int s = 0; s < NSC; ++s) z[s] = nr + (fr - nr) * ((float)s / (float)(NSC-1));
  float zm[63];
  for (int i = 0; i < 63; ++i) zm[i] = 0.5f * (z[i] + z[i+1]);
  float pdf[62]; float sum = 0.f;
  for (int i = 0; i < 62; ++i) { float w = wbuf[r*NSC + 1 + i] + 1e-5f; pdf[i] = w; sum += w; }
  float inv = 1.f / sum;
  float cdf[63]; cdf[0] = 0.f;
  for (int i = 0; i < 62; ++i) cdf[i+1] = cdf[i] + pdf[i] * inv;
  float zs[64];
  for (int j = 0; j < 64; ++j) {
    float u = (float)j / 63.f;
    int ind = 1;
    while (ind < 63 && cdf[ind] <= u) ++ind;     // searchsorted 'right'
    int below = ind - 1;
    int above = (ind < 63) ? ind : 62;
    float cb = cdf[below], ca = cdf[above];
    float bb = zm[below], ba = zm[above];
    float den = (ca - cb < 1e-5f) ? 1.f : (ca - cb);
    zs[j] = bb + ((u - cb) / den) * (ba - bb);
  }
  int i = 0, jj = 0;
  float* out = zf + (size_t)r * NSF;
  for (int k = 0; k < NSF; ++k) {
    bool takeA = (jj >= 64) || (i < 64 && z[i] <= zs[jj]);
    out[k] = takeA ? z[i++] : zs[jj++];
  }
}

extern "C" void kernel_launch(void* const* d_in, const int* in_sizes, int n_in,
                              void* d_out, int out_size, void* d_ws, size_t ws_size,
                              hipStream_t stream) {
  (void)in_sizes; (void)n_in; (void)out_size; (void)ws_size;
  const float* rays = (const float*)d_in[0];
  const float* bc   = (const float*)d_in[1];
  const float* aud  = (const float*)d_in[2];

  // param leaves per net (sorted-key pytree): alpha(W,b), feature(W,b), pts[8](W,b), rgb(W,b), views(W,b)
  PrepArgs pa;
  for (int net = 0; net < 2; ++net) {
    const int bse = 3 + net * 24;
    auto L = [&](int leaf) { return (const float*)d_in[bse + leaf]; };
    for (int l = 0; l < 8; ++l) { pa.W[net*12+l] = L(4 + 2*l); pa.Bv[net*12+l] = L(5 + 2*l); }
    pa.W[net*12+8]  = L(0);  pa.Bv[net*12+8]  = L(1);   // alpha
    pa.W[net*12+9]  = L(2);  pa.Bv[net*12+9]  = L(3);   // feature
    pa.W[net*12+10] = L(22); pa.Bv[net*12+10] = L(23);  // views
    pa.W[net*12+11] = L(20); pa.Bv[net*12+11] = L(21);  // rgb
  }

  char* ws = (char*)d_ws;
  size_t o = 0;
  auto alloc = [&](size_t bytes) { size_t r = o; o += (bytes + 255) & ~(size_t)255; return r; };
  size_t WTC  = alloc((size_t)(WTOT + 4096) * 2);   // +guard for padded N tiles of heads
  size_t WTF  = alloc((size_t)(WTOT + 4096) * 2);
  size_t BSC  = alloc((size_t)BTOT * 4 + 64);
  size_t BSF  = alloc((size_t)BTOT * 4 + 64);
  size_t RAWC = alloc((size_t)NRAYS * NSC * 16);
  size_t RAWF = alloc((size_t)NRAYS * NSF * 16);
  size_t WGT  = alloc((size_t)NRAYS * NSC * 4);
  size_t ZFB  = alloc((size_t)NRAYS * NSF * 4);

  bf16*  wtc   = (bf16*)(ws + WTC);
  bf16*  wtf   = (bf16*)(ws + WTF);
  float* biasc = (float*)(ws + BSC);
  float* biasf = (float*)(ws + BSF);
  float* rawc  = (float*)(ws + RAWC);
  float* rawf  = (float*)(ws + RAWF);
  float* wgt   = (float*)(ws + WGT);
  float* zfb   = (float*)(ws + ZFB);
  float* outp  = (float*)d_out;   // [rgb_map(4096x3) | rgb0(4096x3)]

  {
    long total = 2L * (WTOT + BTOT);
    int blocks = (int)((total + 255) / 256);
    prep_params<<<blocks, 256, 0, stream>>>(pa, wtc, wtf, biasc, biasf);
  }
  size_t shmem = (size_t)(64*400 + 2*64*272) * sizeof(bf16);  // 120832 B

  fused_mlp<<<(NRAYS*NSC)/64, 256, shmem, stream>>>(rays, aud, nullptr, wtc, biasc, rawc, NSC);
  raw2out<<<NRAYS/256, 256, 0, stream>>>(rawc, rays, bc, nullptr, outp + NRAYS*3, wgt, NSC);
  pdf_merge<<<NRAYS/256, 256, 0, stream>>>(rays, wgt, zfb);
  fused_mlp<<<(NRAYS*NSF)/64, 256, shmem, stream>>>(rays, aud, zfb, wtf, biasf, rawf, NSF);
  raw2out<<<NRAYS/256, 256, 0, stream>>>(rawf, rays, bc, zfb, outp, nullptr, NSF);
}
